// WaveEquationPINN_21071109554254
// MI455X (gfx1250) — matrix operand order repaired and checked by reading, not compiled, not run on hardware
//
#include <hip/hip_runtime.h>

typedef __bf16 bf;
typedef __attribute__((ext_vector_type(16))) __bf16 v16bf;
typedef __attribute__((ext_vector_type(8)))  __bf16 v8bf;
typedef __attribute__((ext_vector_type(8)))  float  v8f;

#define HID   512
#define MT    16            // rows per workgroup tile
#define TPB   256           // 8 waves (wave32)
#define SSIZE (MT * HID)    // elements per stream per buffer (8192)

__device__ __forceinline__ v16bf cat8(v8bf lo, v8bf hi) {
    return __builtin_shufflevector(lo, hi, 0,1,2,3,4,5,6,7,8,9,10,11,12,13,14,15);
}

// ---------------------------------------------------------------------------
// Zero the 3 output losses (harness poisons d_out; we must re-zero each call)
// ---------------------------------------------------------------------------
__global__ void wave_pinn_zero(float* out) {
    if (threadIdx.x < 3) out[threadIdx.x] = 0.0f;
}

// ---------------------------------------------------------------------------
// Transpose + bf16-convert W1,W2,W3 (each [K=512][N=512] fp32 row-major)
// into Wt[layer][n][k] bf16 so B-fragment loads are contiguous in K.
// ---------------------------------------------------------------------------
__global__ void wave_pinn_prep(const float* __restrict__ W1,
                               const float* __restrict__ W2,
                               const float* __restrict__ W3,
                               bf* __restrict__ wt) {
    int i = blockIdx.x * TPB + threadIdx.x;      // 0 .. 3*512*512-1
    int l   = i >> 18;                            // layer
    int rem = i & 262143;
    int n   = rem >> 9;
    int k   = rem & (HID - 1);
    const float* W = (l == 0) ? W1 : (l == 1) ? W2 : W3;
    wt[i] = (bf)W[k * HID + n];
}

// ---------------------------------------------------------------------------
// Fused PINN forward+tangent pipeline.
//   NS == 2 : streams {a, a_dot_t}            -> loss_init (out[0]), loss_init_deriv (out[1])
//   NS == 5 : streams {a, a_t, a_tt, a_x, a_xx} -> loss_f (out[2])
// Each block: MT=16 rows; activations live in LDS ping-pong (bf16).
// Layers 1..3 are bf16 WMMA (16x16x32) with f32 accumulate.
// ---------------------------------------------------------------------------
template <int NS>
__global__ __launch_bounds__(TPB)
void wave_pinn_fused(const float* __restrict__ xt,
                     const float* __restrict__ tgt0,   // f (NS=5) or g (NS=2)
                     const float* __restrict__ tgt1,   // g_deriv (NS=2), unused NS=5
                     const float* __restrict__ W0, const float* __restrict__ b0,
                     const bf*    __restrict__ Wt,     // 3 x [512][512] bf16 (transposed)
                     const float* __restrict__ b1,
                     const float* __restrict__ b2,
                     const float* __restrict__ b3,
                     const float* __restrict__ W4, const float* __restrict__ b4,
                     const float* __restrict__ cp,
                     float* __restrict__ out, int nrows)
{
    extern __shared__ char smem[];
    bf* buf0 = (bf*)smem;
    bf* buf1 = buf0 + NS * SSIZE;

    const int tid   = threadIdx.x;
    const int lane  = tid & 31;
    const int wave  = tid >> 5;
    const int rowbase = blockIdx.x * MT;

    // ---------------- layer 0: 2 -> 512 (scalar, tiny) ----------------
    for (int i = tid; i < SSIZE; i += TPB) {
        int r = i >> 9, col = i & (HID - 1);
        float x0 = xt[(rowbase + r) * 2 + 0];
        float x1 = xt[(rowbase + r) * 2 + 1];
        float w0 = W0[col];            // d h / d x
        float w1 = W0[HID + col];      // d h / d t
        float h  = fmaf(x0, w0, fmaf(x1, w1, b0[col]));
        float a  = tanhf(h);
        float s  = 1.0f - a * a;
        buf0[i] = (bf)a;
        if constexpr (NS >= 2) {
            float at = s * w1;                         // first tangent (t)
            buf0[1 * SSIZE + i] = (bf)at;
            if constexpr (NS == 5) {
                buf0[2 * SSIZE + i] = (bf)(-2.0f * a * at * w1);   // second tangent (t), hdd=0
                float ax = s * w0;                                  // first tangent (x)
                buf0[3 * SSIZE + i] = (bf)ax;
                buf0[4 * SSIZE + i] = (bf)(-2.0f * a * ax * w0);   // second tangent (x)
            }
        }
    }
    __syncthreads();

    // ---------------- layers 1..3 : 512 -> 512 via WMMA ----------------
    bf* pin  = buf0;
    bf* pout = buf1;
    const int nbase = wave * 64;        // this wave's 64 output columns
    const int ncol  = lane & 15;        // N within 16-tile (B/C) ; M row (A)
    const int khalf = lane >> 4;        // 0/1 : K-half selector

    for (int l = 0; l < 3; ++l) {
        const bf*    Wl = Wt + l * (HID * HID);
        const float* bl = (l == 0) ? b1 : (l == 1) ? b2 : b3;

        v8f acc[NS][4];
        #pragma unroll
        for (int s = 0; s < NS; ++s)
            #pragma unroll
            for (int nt = 0; nt < 4; ++nt)
                acc[s][nt] = v8f{0.f,0.f,0.f,0.f,0.f,0.f,0.f,0.f};

        #pragma unroll 4
        for (int kb = 0; kb < HID; kb += 32) {
            // B fragments: 4 n-tiles, reused across all NS streams
            v16bf Bf[4];
            #pragma unroll
            for (int nt = 0; nt < 4; ++nt) {
                const bf* p = Wl + (size_t)(nbase + nt * 16 + ncol) * HID + kb + khalf * 8;
                Bf[nt] = cat8(*(const v8bf*)p, *(const v8bf*)(p + 16));
            }
            #pragma unroll
            for (int s = 0; s < NS; ++s) {
                // A fragment from LDS: lane row = ncol, K chunks {kb+koff.., kb+16+koff..}
                const bf* ap = pin + s * SSIZE + ncol * HID + kb + khalf * 8;
                v16bf Af = cat8(*(const v8bf*)ap, *(const v8bf*)(ap + 16));
                #pragma unroll
                for (int nt = 0; nt < 4; ++nt)
                    acc[s][nt] = __builtin_amdgcn_wmma_f32_16x16x32_bf16(
                        false, Af, false, Bf[nt], (short)0, acc[s][nt], false, false);
            }
        }

        // epilogue: bias + tanh + tangent chain rule, write bf16 activations
        #pragma unroll
        for (int nt = 0; nt < 4; ++nt) {
            int col = nbase + nt * 16 + ncol;
            float bb = bl[col];
            #pragma unroll
            for (int i = 0; i < 8; ++i) {
                int m = i + 8 * khalf;           // C layout: M = vgpr + 8*(lane/16)
                float h = acc[0][nt][i] + bb;
                float a = tanhf(h);
                float s = 1.0f - a * a;
                pout[m * HID + col] = (bf)a;
                if constexpr (NS >= 2) {
                    float dh = acc[1][nt][i];
                    float at = s * dh;
                    pout[1 * SSIZE + m * HID + col] = (bf)at;
                    if constexpr (NS == 5) {
                        float ddh = acc[2][nt][i];
                        pout[2 * SSIZE + m * HID + col] = (bf)(s * ddh - 2.0f * a * at * dh);
                        float dhx = acc[3][nt][i];
                        float ax  = s * dhx;
                        pout[3 * SSIZE + m * HID + col] = (bf)ax;
                        float ddx = acc[4][nt][i];
                        pout[4 * SSIZE + m * HID + col] = (bf)(s * ddx - 2.0f * a * ax * dhx);
                    }
                }
            }
        }
        __syncthreads();
        bf* t = pin; pin = pout; pout = t;
    }
    // final activations are in `pin`; `pout` is free scratch

    // ---------------- final layer 512 -> 1 + loss reduction ----------------
    float* red = (float*)pout;                   // >= 512 floats available
    const float scale = 0.5f / (float)nrows;
    int r  = tid >> 4;                           // row 0..15
    int cs = tid & 15;                           // column subset

    if constexpr (NS == 5) {
        float c  = cp[0];
        float c2 = c * c;
        float part = 0.0f;
        for (int col = cs; col < HID; col += 16) {
            float att = (float)pin[2 * SSIZE + r * HID + col];
            float axx = (float)pin[4 * SSIZE + r * HID + col];
            part = fmaf(att - c2 * axx, W4[col], part);
        }
        red[tid] = part;
        __syncthreads();
        if (tid < MT) {
            float p = 0.0f;
            #pragma unroll
            for (int j = 0; j < 16; ++j) p += red[tid * 16 + j];
            float resid = p - tgt0[rowbase + tid];
            atomicAdd(out + 2, scale * resid * resid);
        }
    } else {
        float p0 = 0.0f, p1 = 0.0f;
        for (int col = cs; col < HID; col += 16) {
            float w = W4[col];
            p0 = fmaf((float)pin[r * HID + col],         w, p0);
            p1 = fmaf((float)pin[SSIZE + r * HID + col], w, p1);
        }
        red[tid]       = p0;
        red[256 + tid] = p1;
        __syncthreads();
        if (tid < MT) {
            float u = b4[0], ut = 0.0f;
            #pragma unroll
            for (int j = 0; j < 16; ++j) {
                u  += red[tid * 16 + j];
                ut += red[256 + tid * 16 + j];
            }
            float r0 = u  - tgt0[rowbase + tid];
            float r1 = ut - tgt1[rowbase + tid];
            atomicAdd(out + 0, scale * r0 * r0);
            atomicAdd(out + 1, scale * r1 * r1);
        }
    }
}

// ---------------------------------------------------------------------------
extern "C" void kernel_launch(void* const* d_in, const int* in_sizes, int n_in,
                              void* d_out, int out_size, void* d_ws, size_t ws_size,
                              hipStream_t stream) {
    (void)n_in; (void)out_size; (void)ws_size;
    const float* xt_int  = (const float*)d_in[0];
    const float* f       = (const float*)d_in[1];
    const float* xt_init = (const float*)d_in[2];
    const float* g       = (const float*)d_in[3];
    const float* gd      = (const float*)d_in[4];
    const float* W0 = (const float*)d_in[5];
    const float* b0 = (const float*)d_in[6];
    const float* W1 = (const float*)d_in[7];
    const float* b1 = (const float*)d_in[8];
    const float* W2 = (const float*)d_in[9];
    const float* b2 = (const float*)d_in[10];
    const float* W3 = (const float*)d_in[11];
    const float* b3 = (const float*)d_in[12];
    const float* W4 = (const float*)d_in[13];
    const float* b4 = (const float*)d_in[14];
    const float* c  = (const float*)d_in[15];
    float* out = (float*)d_out;
    bf*    Wt  = (bf*)d_ws;                       // 3 * 512*512 bf16 = 1.5 MB

    const int n_int  = in_sizes[0] / 2;           // 131072
    const int n_init = in_sizes[2] / 2;           // 32768

    wave_pinn_zero<<<1, 32, 0, stream>>>(out);
    wave_pinn_prep<<<(3 * HID * HID) / TPB, TPB, 0, stream>>>(W1, W2, W3, Wt);

    const size_t smem2 = (size_t)2 * 2 * SSIZE * sizeof(bf);   //  64 KB
    const size_t smem5 = (size_t)2 * 5 * SSIZE * sizeof(bf);   // 160 KB (<= 320 KB WGP LDS)

    wave_pinn_fused<2><<<n_init / MT, TPB, smem2, stream>>>(
        xt_init, g, gd, W0, b0, Wt, b1, b2, b3, W4, b4, c, out, n_init);
    wave_pinn_fused<5><<<n_int / MT, TPB, smem5, stream>>>(
        xt_int, f, nullptr, W0, b0, Wt, b1, b2, b3, W4, b4, c, out, n_int);
}
